// EATransformerMHS_39779987096203
// MI455X (gfx1250) — matrix operand order repaired and checked by reading, not compiled, not run on hardware
//
#include <hip/hip_runtime.h>

// MI455X / gfx1250, wave32. Fused kernel:
//   out[b,j,l,i] = sum_k  p[b, k/64, i, j] * (v[b, k/64, j, k%64] + rel[b,i,j,k]) * W[l,k] + bias[l]
// One workgroup per (b,j); 8 waves; each wave computes D(32 l x 32 i) via
// v_wmma_f32_16x16x32_f16 with W (f16, LDS) as A and x^T built in registers as B.
// rel (402 MB, streamed once) is loaded non-temporally to keep L2 for p/v/W/out.

typedef _Float16 v16h __attribute__((ext_vector_type(16)));
typedef _Float16 v8h  __attribute__((ext_vector_type(8)));
typedef _Float16 v4h  __attribute__((ext_vector_type(4)));
typedef float    v8f  __attribute__((ext_vector_type(8)));
typedef float    v4f  __attribute__((ext_vector_type(4)));  // ext-vector: valid for nontemporal builtins

#define BB 2
#define HH 12
#define SS 256
#define EE 64
#define LL 25
#define KK 768      // H*E
#define LPAD 32     // L padded to 2 WMMA M-tiles
#define WPITCH 776  // f16 pitch: 776 halves = 388 dwords == 4 mod 64 banks -> conflict-free
#define KCHUNKS 24  // 768 / 32
#define NTHREADS 256

__global__ __launch_bounds__(NTHREADS)
void mhs_fused_wmma_kernel(const float* __restrict__ p,
                           const float* __restrict__ v,
                           const float* __restrict__ rel,
                           const float* __restrict__ W,
                           const float* __restrict__ bias,
                           float* __restrict__ out)
{
    __shared__ __align__(16) _Float16 lw[LPAD * WPITCH]; // W in f16, rows 25..31 zero
    __shared__ __align__(16) float    lv[KK];            // v[b, :, j, :] flattened over k=h*64+e

    const int tid = threadIdx.x;
    const int b   = blockIdx.y;
    const int j   = blockIdx.x;

    // ---- stage W: zero-pad then convert f32 -> f16 into LDS ----
    {
        uint4 z = make_uint4(0u, 0u, 0u, 0u);
        for (int t = tid; t < (LPAD * WPITCH) / 8; t += NTHREADS)  // 8 halves = 16B
            ((uint4*)lw)[t] = z;
    }
    __syncthreads();
    for (int t = tid; t < LL * (KK / 4); t += NTHREADS) {
        const int l = t / (KK / 4);
        const int c = t % (KK / 4);
        const v4f w = *(const v4f*)(W + (size_t)l * KK + c * 4);
        v4h h;
        h.x = (_Float16)w.x; h.y = (_Float16)w.y;
        h.z = (_Float16)w.z; h.w = (_Float16)w.w;
        *(v4h*)(lw + l * WPITCH + c * 4) = h;
    }
    // ---- stage v[b, :, j, :] into lv ----
    for (int t = tid; t < KK / 4; t += NTHREADS) {
        const int k = t * 4;
        const int h = k >> 6;
        const int e = k & 63;
        *(v4f*)(lv + k) =
            *(const v4f*)(v + (((size_t)b * HH + h) * SS + j) * EE + e);
    }
    __syncthreads();

    const int lane = tid & 31;
    const int wave = tid >> 5;
    const int n    = lane & 15;   // WMMA N index == i offset within tile
    const int hi   = lane >> 4;   // lane half selects K sub-range
    const int i0   = wave * 32;   // 8 waves x 2 N-tiles cover i = 0..255
    const int iA   = i0 + n;      // N-tile 0 row
    const int iB   = i0 + 16 + n; // N-tile 1 row

    // preload attention probs p[b,h,i,j] (L2-resident scalars; h = kc>>1)
    float pvA[HH], pvB[HH];
#pragma unroll
    for (int h = 0; h < HH; ++h) {
        const size_t base = ((size_t)b * HH + h) * SS;
        pvA[h] = p[(base + iA) * SS + j];
        pvB[h] = p[(base + iB) * SS + j];
    }

    const float* relrowA = rel + (((size_t)b * SS + iA) * SS + j) * KK;
    const float* relrowB = rel + (((size_t)b * SS + iB) * SS + j) * KK;

    v8f acc00 = {};  // l 0..15  x iA-tile
    v8f acc01 = {};  // l 0..15  x iB-tile
    v8f acc10 = {};  // l 16..31 x iA-tile
    v8f acc11 = {};  // l 16..31 x iB-tile

#pragma unroll
    for (int kc = 0; kc < KCHUNKS; ++kc) {
        const int k16 = kc * 32 + hi * 16;
        const int ka  = kc * 32 + hi * 16;

        // A fragments from LDS first (latency covered by the VALU chain below)
        const v8h a0lo = *(const v8h*)(lw + n * WPITCH + ka);
        const v8h a0hi = *(const v8h*)(lw + n * WPITCH + ka + 8);
        const v8h a1lo = *(const v8h*)(lw + (16 + n) * WPITCH + ka);
        const v8h a1hi = *(const v8h*)(lw + (16 + n) * WPITCH + ka + 8);
        const v16h a0 = __builtin_shufflevector(a0lo, a0hi,
                        0,1,2,3,4,5,6,7,8,9,10,11,12,13,14,15);
        const v16h a1 = __builtin_shufflevector(a1lo, a1hi,
                        0,1,2,3,4,5,6,7,8,9,10,11,12,13,14,15);

        const float psA = pvA[kc >> 1];
        const float psB = pvB[kc >> 1];

        // B fragments: 16 consecutive k of rows iA/iB, x = p*(v+rel) in f16.
        // rel is streamed once -> non-temporal loads; v row shared from LDS.
        v16h bfA, bfB;
#pragma unroll
        for (int q = 0; q < 4; ++q) {
            const v4f vv = *(const v4f*)(lv + k16 + q * 4);
            const v4f rA = __builtin_nontemporal_load(
                               (const v4f*)(relrowA + k16 + q * 4));
            const v4f rB = __builtin_nontemporal_load(
                               (const v4f*)(relrowB + k16 + q * 4));
            bfA[q * 4 + 0] = (_Float16)(psA * (vv.x + rA.x));
            bfA[q * 4 + 1] = (_Float16)(psA * (vv.y + rA.y));
            bfA[q * 4 + 2] = (_Float16)(psA * (vv.z + rA.z));
            bfA[q * 4 + 3] = (_Float16)(psA * (vv.w + rA.w));
            bfB[q * 4 + 0] = (_Float16)(psB * (vv.x + rB.x));
            bfB[q * 4 + 1] = (_Float16)(psB * (vv.y + rB.y));
            bfB[q * 4 + 2] = (_Float16)(psB * (vv.z + rB.z));
            bfB[q * 4 + 3] = (_Float16)(psB * (vv.w + rB.w));
        }

        acc00 = __builtin_amdgcn_wmma_f32_16x16x32_f16(false, a0, false, bfA,
                                                       (short)0, acc00, false, false);
        acc10 = __builtin_amdgcn_wmma_f32_16x16x32_f16(false, a1, false, bfA,
                                                       (short)0, acc10, false, false);
        acc01 = __builtin_amdgcn_wmma_f32_16x16x32_f16(false, a0, false, bfB,
                                                       (short)0, acc01, false, false);
        acc11 = __builtin_amdgcn_wmma_f32_16x16x32_f16(false, a1, false, bfB,
                                                       (short)0, acc11, false, false);
    }

    // ---- store D + bias -> out[b, j, l, i]; lanes contiguous along i ----
    float* obase = out + ((size_t)b * SS + j) * (LL * SS);
#pragma unroll
    for (int g = 0; g < 8; ++g) {
        const int l0 = hi * 8 + g;          // 0..15, always < 25
        const float bias0 = bias[l0];
        obase[l0 * SS + iA] = acc00[g] + bias0;
        obase[l0 * SS + iB] = acc01[g] + bias0;
        const int l1 = 16 + hi * 8 + g;     // 16..31
        if (l1 < LL) {
            const float bias1 = bias[l1];
            obase[l1 * SS + iA] = acc10[g] + bias1;
            obase[l1 * SS + iB] = acc11[g] + bias1;
        }
    }
}

extern "C" void kernel_launch(void* const* d_in, const int* in_sizes, int n_in,
                              void* d_out, int out_size, void* d_ws, size_t ws_size,
                              hipStream_t stream) {
    const float* p    = (const float*)d_in[0];  // attention_probs  [B,H,S,S]
    const float* v    = (const float*)d_in[1];  // value_attentions [B,H,S,E]
    const float* rel  = (const float*)d_in[2];  // relative_positions [B,S,S,H*E]
    const float* W    = (const float*)d_in[3];  // [L, H*E]
    const float* bias = (const float*)d_in[4];  // [L]
    float* out = (float*)d_out;                 // [B, S, L, S]

    dim3 grid(SS, BB);   // one workgroup per (j, b)
    mhs_fused_wmma_kernel<<<grid, NTHREADS, 0, stream>>>(p, v, rel, W, bias, out);
}
